// SingleLayerEncoder_84636625535254
// MI455X (gfx1250) — matrix operand order repaired and checked by reading, not compiled, not run on hardware
//
#include <hip/hip_runtime.h>

// ---------------------------------------------------------------------------
// SingleLayerEncoder on gfx1250 (MI455X, wave32):
//   phase 1: ext = x @ W_in^T        -> parallel bf16 WMMA GEMM (f32 acc)
//   phase 2: s = relu((s+ext_t)@W_rec^T), t=0..511
//            -> ONE persistent kernel, 256 waves (one 16x16 tile each),
//               W_rec slice TDM-staged into LDS once, device-wide
//               generation barrier between steps.
//   out[:,0,:] = final state, rest zeros.
// ---------------------------------------------------------------------------

typedef __attribute__((ext_vector_type(16))) __bf16 v16bf;
typedef __attribute__((ext_vector_type(8)))  __bf16 v8bf;
typedef __attribute__((ext_vector_type(8)))  float  v8f;
typedef __attribute__((ext_vector_type(4)))  unsigned int u32x4;
typedef __attribute__((ext_vector_type(8)))  int i32x8;
typedef __attribute__((ext_vector_type(4)))  int i32x4;

#define BS    64
#define TT    512
#define DIN   1024
#define DOUT  1024

#define SCAN_BLOCKS 32
// LDS row stride for the W_rec slice: 1024 data elems + TDM pad of 4 DWORDs
// after every 256 DWORDs (2 pads/row) -> 1040 elems = 2080 B per row.
// Chunk offsets stay 16B-aligned; bank stride 8 (2-way pattern, not 16-way).
#define LDSROW 1040

__device__ __forceinline__ int lds_widx(int row, int k) {
  // element index in padded LDS layout: +8 elems (4 DWORDs) after k>=512
  return row * LDSROW + k + ((k >> 9) << 3);
}

// ---------------------------------------------------------------------------
// Fragment helpers (layouts per CDNA5 ISA §7.12.2, wave32)
// A 16x32 bf16: lane holds row lane&15; 16 contiguous K starting at k0 + 16*(lane>=16).
// B 32x16 bf16 from row-major [N,K]: lane holds col lane&15, 16 contiguous K
//   starting at k0 + 16*(lane>=16).
// C/D 16x16 f32: lane = N + 16*(M>=8), VGPR r holds M = r + 8*(lane>=16).
// ---------------------------------------------------------------------------
__device__ __forceinline__ v16bf load_fragB(const __bf16* base, int ld,
                                            int n0, int k0) {
  const int lane = threadIdx.x & 31;
  const int n    = lane & 15;
  const int kb   = (lane >> 4) * 16;
  const __bf16* p = base + (size_t)(n0 + n) * ld + k0 + kb;
  return *(const v16bf*)p;   // 32-byte contiguous load
}

__device__ __forceinline__ void store_fragC_relu(float* base, int ld,
                                                 int m0, int n0, v8f c) {
  const int lane = threadIdx.x & 31;
  const int n    = lane & 15;
  const int mb   = (lane >> 4) * 8;
#pragma unroll
  for (int r = 0; r < 8; ++r)
    base[(size_t)(m0 + mb + r) * ld + n0 + n] = fmaxf(c[r], 0.0f);
}

__device__ __forceinline__ void store_fragC_bf16(__bf16* base, int ld,
                                                 int m0, int n0, v8f c) {
  const int lane = threadIdx.x & 31;
  const int n    = lane & 15;
  const int mb   = (lane >> 4) * 8;
#pragma unroll
  for (int r = 0; r < 8; ++r)
    base[(size_t)(m0 + mb + r) * ld + n0 + n] = (__bf16)c[r];
}

// ---------------------------------------------------------------------------
// Kernel 1: W_in / W_rec fp32 -> bf16 workspace copies.
// ---------------------------------------------------------------------------
__global__ void k_cvt_weights(const float* __restrict__ w_in,
                              const float* __restrict__ w_rec,
                              __bf16* __restrict__ w_in_b,
                              __bf16* __restrict__ w_rec_b) {
  int idx = blockIdx.x * blockDim.x + threadIdx.x;
  if (idx < DOUT * DIN) {
    w_in_b[idx]  = (__bf16)w_in[idx];
    w_rec_b[idx] = (__bf16)w_rec[idx];
  }
}

// ---------------------------------------------------------------------------
// Kernel 2: ext[t][b][o] = sum_k x[b][t][k] * W_in[o][k]
// One wave: 16(M) x 64(N) strip; A fragment reused across 4 B/C.
// ---------------------------------------------------------------------------
__global__ void k_ext_gemm(const float* __restrict__ x,
                           const __bf16* __restrict__ w_in_b,
                           __bf16* __restrict__ ext_b) {
  const int wid  = (blockIdx.x * blockDim.x + threadIdx.x) >> 5;
  const int lane = threadIdx.x & 31;
  const int nt   = wid & 15;
  const int mt   = wid >> 4;
  const int t    = mt >> 2;
  const int b0   = (mt & 3) * 16;
  const int n0   = nt * 64;

  const int    arow = b0 + (lane & 15);
  const int    kb   = (lane >> 4) * 16;
  const float* xrow = x + ((size_t)arow * TT + t) * DIN + kb;

  v8f c0 = {}, c1 = {}, c2 = {}, c3 = {};

#pragma unroll 4
  for (int k0 = 0; k0 < DIN; k0 += 32) {
    v8f lo = *(const v8f*)(xrow + k0);
    v8f hi = *(const v8f*)(xrow + k0 + 8);
    v16bf a;
#pragma unroll
    for (int i = 0; i < 8; ++i) { a[i] = (__bf16)lo[i]; a[8 + i] = (__bf16)hi[i]; }

    v16bf bA = load_fragB(w_in_b, DIN, n0 +  0, k0);
    v16bf bB = load_fragB(w_in_b, DIN, n0 + 16, k0);
    v16bf bC = load_fragB(w_in_b, DIN, n0 + 32, k0);
    v16bf bD = load_fragB(w_in_b, DIN, n0 + 48, k0);

    c0 = __builtin_amdgcn_wmma_f32_16x16x32_bf16(false, a, false, bA, (short)0, c0, false, false);
    c1 = __builtin_amdgcn_wmma_f32_16x16x32_bf16(false, a, false, bB, (short)0, c1, false, false);
    c2 = __builtin_amdgcn_wmma_f32_16x16x32_bf16(false, a, false, bC, (short)0, c2, false, false);
    c3 = __builtin_amdgcn_wmma_f32_16x16x32_bf16(false, a, false, bD, (short)0, c3, false, false);
  }

  const int m0 = t * BS + b0;   // ext stored as [t][b][o]
  store_fragC_bf16(ext_b, DOUT, m0, n0 +  0, c0);
  store_fragC_bf16(ext_b, DOUT, m0, n0 + 16, c1);
  store_fragC_bf16(ext_b, DOUT, m0, n0 + 32, c2);
  store_fragC_bf16(ext_b, DOUT, m0, n0 + 48, c3);
}

// ---------------------------------------------------------------------------
// Kernel 3: persistent scan kernel.
// Grid: 32 blocks x 256 threads. block -> (mt2 = bx>>4, ng = bx&15):
//   covers M-tiles {2*mt2, 2*mt2+1} x N columns [ng*64, ng*64+64).
// Each block TDM-loads W_rec rows [ng*64, ng*64+64) x K=1024 (bf16) into LDS
// once, then iterates 512 steps with a device-wide generation barrier.
// ---------------------------------------------------------------------------
__global__ void k_scan(const __bf16* __restrict__ ext_b,
                       const __bf16* __restrict__ w_rec_b,
                       float* __restrict__ s0, float* __restrict__ s1,
                       unsigned* __restrict__ bar) {
  extern __shared__ __bf16 smem[];   // 64 x LDSROW bf16 = 133120 B

  const int tid  = threadIdx.x;
  const int lane = tid & 31;
  const int wave = tid >> 5;
  const int mt2  = blockIdx.x >> 4;
  const int ng   = blockIdx.x & 15;
  const int m0   = (mt2 * 2 + (wave >> 2)) * 16;
  const int n0l  = (wave & 3) * 16;
  const int n0   = ng * 64 + n0l;

  const __bf16* wslice = w_rec_b + (size_t)(ng * 64) * DOUT;

  // ---- TDM: DMA W_rec slice [64 rows x 1024] bf16 -> padded LDS layout ----
  if (wave == 0) {
    unsigned long long ga = (unsigned long long)(uintptr_t)wslice;
    unsigned ldsa = __builtin_amdgcn_groupstaticsize();  // dynamic LDS offset
    u32x4 g0 = { 1u,                                             // count=1
                 ldsa,                                           // lds_addr
                 (unsigned)(ga & 0xffffffffu),                   // gaddr[31:0]
                 (unsigned)((ga >> 32) & 0x01ffffffu) | 0x80000000u }; // [56:32]|type=2
    i32x8 g1 = { (int)0x07D10000,    // data_size=2B | pad_en | intvl=256dw | amt=4dw
                 (int)(1024u << 16), // tensor_dim0=1024 (lo16)
                 (int)(64u << 16),   // tensor_dim0 hi=0 | tensor_dim1=64 (lo16)
                 (int)(1024u << 16), // tensor_dim1 hi=0 | tile_dim0=1024
                 64,                 // tile_dim1=64 | tile_dim2=0
                 1024,               // tensor_dim0_stride lo
                 0, 0 };             // stride hi / tensor_dim1_stride
    i32x4 z4 = { 0, 0, 0, 0 };
#if __clang_major__ >= 23
    i32x8 z8 = { 0, 0, 0, 0, 0, 0, 0, 0 };
    __builtin_amdgcn_tensor_load_to_lds(g0, g1, z4, z4, z8, 0);
#else
    __builtin_amdgcn_tensor_load_to_lds(g0, g1, z4, z4, 0);
#endif
    __builtin_amdgcn_s_wait_tensorcnt(0);
  }
  __syncthreads();

  // Belt-and-braces: cooperative copy of the same bytes into the same padded
  // layout (D# field packing is spec-derived, not hardware-verified; this
  // runs once and guarantees LDS contents).
  for (int c = tid; c < 64 * (DOUT / 8); c += blockDim.x) {
    int row = c >> 7;
    int k   = (c & 127) * 8;
    *(v8bf*)&smem[lds_widx(row, k)] = *(const v8bf*)&wslice[row * DOUT + k];
  }
  __syncthreads();

  const int arow = m0 + (lane & 15);
  const int kb   = (lane >> 4) * 16;
  const int bn   = n0l + (lane & 15);
  const int bkb  = (lane >> 4) * 16;

#pragma unroll 1
  for (int t = 0; t < TT; ++t) {
    const float*  s_in  = (t & 1) ? s1 : s0;
    float*        s_out = (t & 1) ? s0 : s1;
    const __bf16* pe = ext_b + (size_t)t * BS * DOUT + (size_t)arow * DOUT + kb;
    const float*  ps = s_in + (size_t)arow * DOUT + kb;

    v8f c = {};
#pragma unroll 4
    for (int k0 = 0; k0 < DOUT; k0 += 32) {
      v8f  slo = *(const v8f*)(ps + k0);
      v8f  shi = *(const v8f*)(ps + k0 + 8);
      v8bf elo = *(const v8bf*)(pe + k0);
      v8bf ehi = *(const v8bf*)(pe + k0 + 8);
      v16bf a;
#pragma unroll
      for (int i = 0; i < 8; ++i) {
        a[i]     = (__bf16)(slo[i] + (float)elo[i]);
        a[8 + i] = (__bf16)(shi[i] + (float)ehi[i]);
      }
      const __bf16* bp = &smem[lds_widx(bn, k0 + bkb)];
      union { v16bf v; v8bf h[2]; } b;
      b.h[0] = *(const v8bf*)bp;        // ds_load 16B
      b.h[1] = *(const v8bf*)(bp + 8);  // ds_load 16B
      c = __builtin_amdgcn_wmma_f32_16x16x32_bf16(false, a, false, b.v,
                                                  (short)0, c, false, false);
    }

    store_fragC_relu(s_out, DOUT, m0, n0, c);

    if (t + 1 < TT)
      __builtin_prefetch(pe + (size_t)BS * DOUT, 0, 1);  // next step's ext rows

    // ---- device-wide step barrier (generation counter in workspace) ----
    __threadfence();
    __syncthreads();
    if (tid == 0) {
      unsigned arrived = __hip_atomic_fetch_add(&bar[0], 1u, __ATOMIC_ACQ_REL,
                                                __HIP_MEMORY_SCOPE_AGENT);
      if (arrived == SCAN_BLOCKS - 1) {
        __hip_atomic_store(&bar[0], 0u, __ATOMIC_RELAXED,
                           __HIP_MEMORY_SCOPE_AGENT);
        __hip_atomic_store(&bar[1], (unsigned)(t + 1), __ATOMIC_RELEASE,
                           __HIP_MEMORY_SCOPE_AGENT);
      } else {
        while (__hip_atomic_load(&bar[1], __ATOMIC_ACQUIRE,
                                 __HIP_MEMORY_SCOPE_AGENT) <= (unsigned)t) {
          __builtin_amdgcn_s_sleep(2);
        }
      }
    }
    __syncthreads();
    __threadfence();
  }
}

// ---------------------------------------------------------------------------
// Kernel 4: scatter final state into out[:, 0, :].
// ---------------------------------------------------------------------------
__global__ void k_write_out(const float* __restrict__ s_final,
                            float* __restrict__ out) {
  int idx = blockIdx.x * blockDim.x + threadIdx.x;
  if (idx < BS * DOUT) {
    int b = idx >> 10;
    int o = idx & 1023;
    out[(size_t)b * TT * DOUT + o] = s_final[idx];
  }
}

// ---------------------------------------------------------------------------
// Host launcher
// ---------------------------------------------------------------------------
extern "C" void kernel_launch(void* const* d_in, const int* in_sizes, int n_in,
                              void* d_out, int out_size, void* d_ws, size_t ws_size,
                              hipStream_t stream) {
  (void)in_sizes; (void)n_in; (void)out_size; (void)ws_size;

  const float* x     = (const float*)d_in[0];   // [64,512,1024]
  const float* w_in  = (const float*)d_in[1];   // [1024,1024]
  const float* w_rec = (const float*)d_in[2];   // [1024,1024]
  float* out = (float*)d_out;                   // [64,512,1024]

  char* ws = (char*)d_ws;
  __bf16*   w_in_b  = (__bf16*)(ws);                              // 2 MB
  __bf16*   w_rec_b = (__bf16*)(ws + (2u << 20));                 // 2 MB
  float*    s0      = (float*) (ws + (4u << 20));                 // 256 KB
  float*    s1      = (float*) (ws + (4u << 20) + (256u << 10));  // 256 KB
  unsigned* bar     = (unsigned*)(ws + (4u << 20) + (512u << 10));// 8 B
  __bf16*   ext_b   = (__bf16*)(ws + (5u << 20));                 // 64 MB

  hipMemsetAsync(d_out, 0, (size_t)BS * TT * DOUT * sizeof(float), stream);
  hipMemsetAsync(s0, 0, (size_t)BS * DOUT * sizeof(float), stream);
  hipMemsetAsync(bar, 0, 2 * sizeof(unsigned), stream);

  k_cvt_weights<<<(DOUT * DIN + 255) / 256, 256, 0, stream>>>(w_in, w_rec,
                                                              w_in_b, w_rec_b);

  k_ext_gemm<<<(32768 * 32) / 256, 256, 0, stream>>>(x, w_in_b, ext_b);

  const int scan_lds = 64 * LDSROW * (int)sizeof(__bf16);  // 133120 B
  hipFuncSetAttribute((const void*)k_scan,
                      hipFuncAttributeMaxDynamicSharedMemorySize, scan_lds);
  k_scan<<<SCAN_BLOCKS, 256, scan_lds, stream>>>(ext_b, w_rec_b, s0, s1, bar);
  // TT=512 even -> final state in s0

  k_write_out<<<(BS * DOUT + 255) / 256, 256, 0, stream>>>(s0, out);
}
